// Decoder_57432302682534
// MI455X (gfx1250) — matrix operand order repaired and checked by reading, not compiled, not run on hardware
//
#include <hip/hip_runtime.h>
#include <hip/hip_bf16.h>
#include <stdint.h>

typedef __attribute__((ext_vector_type(16))) _Float16 v16h;
typedef __attribute__((ext_vector_type(8)))  float    v8f;
typedef __attribute__((ext_vector_type(4)))  int      v4i;

#define B_N   32768
#define L_N   7
#define T_N   6
#define NEGV  (-1000000000.0f)

#define ASTR 520    // halves, padded stride for 16x512 f16 A buffer (bank-conflict-free b128 reads)
#define ESTR 264    // halves, padded stride for 16x256 f16 buffers
#define SSTR 260    // floats, padded stride for 16x256 f32 st buffer

#if defined(__gfx1250__) && __has_builtin(__builtin_amdgcn_global_load_async_to_lds_b128)
#define HAVE_ASYNC 1
#else
#define HAVE_ASYNC 0
#endif

union AFragU { v16h v; uint4 q[2]; };
union H4 { uint2 u2; _Float16 h[4]; };

// ---- helpers ----------------------------------------------------------------
static __device__ __forceinline__ float fast_tanh(float x) {
#if defined(__gfx1250__) && __has_builtin(__builtin_amdgcn_tanhf)
  return __builtin_amdgcn_tanhf(x);
#else
  float e = __expf(2.f * x);
  return 1.f - 2.f / (e + 1.f);
#endif
}

// 16B-per-lane async copy global->LDS (raw bytes), ASYNCcnt-tracked.
static __device__ __forceinline__ void copy_b128_to_lds(const _Float16* g, _Float16* l) {
#if HAVE_ASYNC
  __builtin_amdgcn_global_load_async_to_lds_b128(
      (__attribute__((address_space(1))) v4i*)(uintptr_t)g,
      (__attribute__((address_space(3))) v4i*)l, 0, 0);
#else
  *(uint4*)l = *(const uint4*)g;
#endif
}
static __device__ __forceinline__ void async_copy_wait() {
#if HAVE_ASYNC
#if __has_builtin(__builtin_amdgcn_s_wait_asynccnt)
  __builtin_amdgcn_s_wait_asynccnt(0);
#else
  asm volatile("s_wait_asynccnt 0" ::: "memory");
#endif
#endif
}

// B fragment: pre-packed per-lane contiguous 16 halves (32B) in global scratch.
static __device__ __forceinline__ v16h loadB(const _Float16* pack, int nt, int kt, int KT, int lane) {
  return *(const v16h*)(pack + (((size_t)(nt * KT + kt)) * 32 + (size_t)lane) * 16);
}
// A fragment from row-major f16 LDS buffer, contiguous K order (matches packed B):
// lanes 0-15: V0..7 = K[0..15]; lanes 16-31: K[16..31].
static __device__ __forceinline__ v16h loadA_lds(const _Float16* buf, int strideH, int kt, int lane) {
  int m  = lane & 15;
  int hs = (lane >> 4) * 16;
  AFragU a;
  a.q[0] = *(const uint4*)(buf + m * strideH + kt * 32 + hs);
  a.q[1] = *(const uint4*)(buf + m * strideH + kt * 32 + 8 + hs);
  return a.v;
}
static __device__ __forceinline__ v8f wmma_f16(v16h a, v16h b, v8f c) {
  return __builtin_amdgcn_wmma_f32_16x16x32_f16(false, a, false, b, (short)0, c, false, false);
}

// ---- f32 -> f16 bulk conversion of encoder_output ---------------------------
__global__ __launch_bounds__(256)
void cvt_kernel(const float* __restrict__ src, _Float16* __restrict__ dst, int n4) {
  int stride = gridDim.x * blockDim.x;
  for (int i = blockIdx.x * blockDim.x + threadIdx.x; i < n4; i += stride) {
    float4 f = ((const float4*)src)[i];
    H4 p;
    p.h[0] = (_Float16)f.x; p.h[1] = (_Float16)f.y; p.h[2] = (_Float16)f.z; p.h[3] = (_Float16)f.w;
    ((uint2*)dst)[i] = p.u2;
  }
}

// ---- weight packing: B[k][n] = W[n][k], packed into per-lane fragment order --
__global__ __launch_bounds__(256) void pack_b_kernel(_Float16* __restrict__ dst,
                                                     const float* __restrict__ src,
                                                     int KT, int koff) {
  int g = blockIdx.x * blockDim.x + threadIdx.x;     // 0 .. 16*8*32-1
  if (g >= 16 * 8 * 32) return;
  int lane = g & 31;
  int kt   = (g >> 5) & 7;
  int nt   = g >> 8;
  int n  = nt * 16 + (lane & 15);
  int kb = kt * 32 + (lane >> 4) * 16;
  _Float16* out = dst + (((size_t)(nt * KT + koff + kt)) * 32 + (size_t)lane) * 16;
#pragma unroll
  for (int j = 0; j < 8; ++j) {
    out[2 * j]     = (_Float16)src[(size_t)n * 256 + kb + 2 * j];
    out[2 * j + 1] = (_Float16)src[(size_t)n * 256 + kb + 2 * j + 1];
  }
}

// ---- fused scan (5 live steps) + qt + inp -----------------------------------
__global__ __launch_bounds__(128)
void scan_kernel(const _Float16* __restrict__ enc16, const long long* __restrict__ xes,
                 const _Float16* __restrict__ packSel, const _Float16* __restrict__ packEdge,
                 const _Float16* __restrict__ packIn, const float* __restrict__ b_in,
                 float* __restrict__ inp_out) {
  __shared__ _Float16 Abuf[16 * ASTR];   // 16 x 512 f16 : [enc_h | enc_v]
  __shared__ _Float16 Ebuf[16 * ESTR];   // edge (later qt) as f16 A operand
  __shared__ float    Sbuf[16 * SSTR];   // subtree carry f32
  __shared__ unsigned sHE[16], sVE[16], sCOND[16];

  const int tid  = threadIdx.x;
  const int lane = tid & 31;
  const int wv   = tid >> 5;
  const int ntb  = wv * 4;               // each wave owns 4 N-tiles (64 columns)
  const int b0   = blockIdx.x * 16;
  const int n0   = lane & 15;
  const int mh   = (lane >> 4) * 8;

  for (int i = tid; i < 16 * SSTR; i += 128) Sbuf[i] = 0.f;

  v8f qacc[4] = {};

  for (int t = 0; t < 5; ++t) {
    if (tid < 16) {
      int m = tid;
      size_t xb = ((size_t)(b0 + m) * T_N + t) * 3;
      long long h = xes[xb + 0], v = xes[xb + 1], tt = xes[xb + 2];
      sHE[m]   = ((unsigned)(b0 + m) * 7u + (unsigned)h) * 256u;
      sVE[m]   = ((unsigned)(b0 + m) * 7u + (unsigned)v) * 256u;
      sCOND[m] = (tt == 0) ? 1u : 0u;
    }
    __syncthreads();

    // Stage A: 32 row-copies (16 enc_h + 16 enc_v), each 512B = 32 lanes x 16B.
    for (int rr = wv; rr < 32; rr += 4) {
      int m = rr & 15;
      unsigned srcOff = (rr < 16) ? sHE[m] : sVE[m];
      int dhalf = m * ASTR + ((rr < 16) ? 0 : 256);
      copy_b128_to_lds(enc16 + srcOff + lane * 8, &Abuf[dhalf + lane * 8]);
    }
    // Prefetch next step's gather rows while this step computes.
    if (t < 4) {
      int m = tid >> 3, seg = tid & 7;                  // 16 rows x 8 x 64B
      size_t xb = ((size_t)(b0 + m) * T_N + (t + 1)) * 3;
      unsigned hh = (unsigned)xes[xb + 0], vv = (unsigned)xes[xb + 1];
      __builtin_prefetch(enc16 + ((unsigned)(b0 + m) * 7u + hh) * 256u + seg * 32, 0, 0);
      __builtin_prefetch(enc16 + ((unsigned)(b0 + m) * 7u + vv) * 256u + seg * 32, 0, 0);
    }
    async_copy_wait();
    __syncthreads();

    // Both edge branches: e0 = A@[W_h;W_v].T, e1 = A@[Ws_h;Ws_v].T  (K=512)
    v8f e0[4] = {}, e1[4] = {};
    for (int kt = 0; kt < 16; ++kt) {
      v16h a = loadA_lds(Abuf, ASTR, kt, lane);
#pragma unroll
      for (int j = 0; j < 4; ++j) {
        v16h bb0 = loadB(packSel, ntb + j, kt,      32, lane);
        v16h bb1 = loadB(packSel, ntb + j, kt + 16, 32, lane);
        e0[j] = wmma_f16(a, bb0, e0[j]);
        e1[j] = wmma_f16(a, bb1, e1[j]);
      }
    }
    // Per-row branch select, stage edge as f16 A operand.
    int cnd[8];
#pragma unroll
    for (int r = 0; r < 8; ++r) cnd[r] = (int)sCOND[mh + r];
    v8f esel[4];
#pragma unroll
    for (int j = 0; j < 4; ++j) {
      int n = (ntb + j) * 16 + n0;
#pragma unroll
      for (int r = 0; r < 8; ++r) {
        float ev = cnd[r] ? e0[j][r] : e1[j][r];
        esel[j][r] = ev;
        Ebuf[(mh + r) * ESTR + n] = (_Float16)ev;
      }
    }
    __syncthreads();

    // subtree = max(st, edge @ W_edge.T)
    v8f sacc[4] = {};
    for (int kt = 0; kt < 8; ++kt) {
      v16h a = loadA_lds(Ebuf, ESTR, kt, lane);
#pragma unroll
      for (int j = 0; j < 4; ++j) {
        v16h b = loadB(packEdge, ntb + j, kt, 8, lane);
        sacc[j] = wmma_f16(a, b, sacc[j]);
      }
    }
#pragma unroll
    for (int j = 0; j < 4; ++j) {
      int n = (ntb + j) * 16 + n0;
#pragma unroll
      for (int r = 0; r < 8; ++r) {
        int m = mh + r;
        float sn = fmaxf(Sbuf[m * SSTR + n], sacc[j][r]);
        if (t == 4) qacc[j][r] = fmaxf(0.f, esel[j][r] + sn);  // qt = relu(el4 + st4)
        else        Sbuf[m * SSTR + n] = sn;
      }
    }
    __syncthreads();
  }

  // stage qt as f16 A operand (reuse Ebuf)
#pragma unroll
  for (int j = 0; j < 4; ++j) {
    int n = (ntb + j) * 16 + n0;
#pragma unroll
    for (int r = 0; r < 8; ++r)
      Ebuf[(mh + r) * ESTR + n] = (_Float16)qacc[j][r];
  }
  __syncthreads();

  // inp = qt @ W_in.T + b_in
  v8f iacc[4] = {};
  for (int kt = 0; kt < 8; ++kt) {
    v16h a = loadA_lds(Ebuf, ESTR, kt, lane);
#pragma unroll
    for (int j = 0; j < 4; ++j) {
      v16h b = loadB(packIn, ntb + j, kt, 8, lane);
      iacc[j] = wmma_f16(a, b, iacc[j]);
    }
  }
#pragma unroll
  for (int j = 0; j < 4; ++j) {
    int n = (ntb + j) * 16 + n0;
    float bi = b_in[n];
#pragma unroll
    for (int r = 0; r < 8; ++r)
      inp_out[(size_t)(b0 + mh + r) * 256 + n] = iacc[j][r] + bi;
  }
}

// ---- ctx GEMM + tanh + V-dot + mask + 10*tanh -------------------------------
__global__ __launch_bounds__(128)
void attn_kernel(const _Float16* __restrict__ enc16, const _Float16* __restrict__ packCtx,
                 const float* __restrict__ inp_g, const float* __restrict__ b_ctx,
                 const float* __restrict__ V, const int* __restrict__ mask,
                 float* __restrict__ att_g) {
  __shared__ _Float16 X[4][16 * ESTR];
  const int tid = threadIdx.x, lane = tid & 31, wv = tid >> 5;
  const size_t r0 = ((size_t)blockIdx.x * 4 + wv) * 16;   // rows = flattened (b,l)
  const int n0 = lane & 15;
  const int mh = (lane >> 4) * 8;

  // rows of reshaped (B*L, 256) f16 encoder are contiguous: async row copies
#pragma unroll
  for (int m = 0; m < 16; ++m)
    copy_b128_to_lds(enc16 + (r0 + m) * 256 + lane * 8, &X[wv][m * ESTR + lane * 8]);
  async_copy_wait();
  __syncthreads();

  int brow[8];
#pragma unroll
  for (int r = 0; r < 8; ++r) brow[r] = (int)((r0 + mh + r) / 7);

  float rs[8] = {0.f, 0.f, 0.f, 0.f, 0.f, 0.f, 0.f, 0.f};
  for (int nt = 0; nt < 16; ++nt) {
    v8f acc = {};
#pragma unroll
    for (int kt = 0; kt < 8; ++kt) {
      v16h a = loadA_lds(X[wv], ESTR, kt, lane);
      v16h b = loadB(packCtx, nt, kt, 8, lane);
      acc = wmma_f16(a, b, acc);
    }
    int n = nt * 16 + n0;
    float bc = b_ctx[n], Vn = V[n];
#pragma unroll
    for (int r = 0; r < 8; ++r) {
      float ip = inp_g[(size_t)brow[r] * 256 + n];
      rs[r] += Vn * fast_tanh(acc[r] + bc + ip);
    }
  }
  // reduce across the 16 lanes that share rows (n within tile)
#pragma unroll
  for (int off = 1; off < 16; off <<= 1)
#pragma unroll
    for (int r = 0; r < 8; ++r) rs[r] += __shfl_xor(rs[r], off, 32);

  if (n0 == 0) {
#pragma unroll
    for (int r = 0; r < 8; ++r) {
      size_t row = r0 + mh + r;
      float a = mask[row] ? rs[r] : NEGV;
      att_g[row] = 10.f * fast_tanh(a);
    }
  }
}

// ---- per-column (axis=0) softmax denominators; att in [-10,10] so max=10 ----
__global__ void zero_denom(float* d) { if (threadIdx.x < 8) d[threadIdx.x] = 0.f; }

__global__ __launch_bounds__(256)
void denom_kernel(const float* __restrict__ att_g, float* __restrict__ denom) {
  __shared__ float red[256];
  float s[7] = {0.f, 0.f, 0.f, 0.f, 0.f, 0.f, 0.f};
  int stride = gridDim.x * blockDim.x;
  for (int b = blockIdx.x * blockDim.x + threadIdx.x; b < B_N; b += stride)
#pragma unroll
    for (int l = 0; l < 7; ++l) s[l] += __expf(att_g[(size_t)b * 7 + l] - 10.f);
  for (int l = 0; l < 7; ++l) {
    red[threadIdx.x] = s[l];
    __syncthreads();
    for (int o = 128; o > 0; o >>= 1) {
      if (threadIdx.x < o) red[threadIdx.x] += red[threadIdx.x + o];
      __syncthreads();
    }
    if (threadIdx.x == 0) atomicAdd(&denom[l], red[0]);
    __syncthreads();
  }
}

// ---- Threefry2x32 + Gumbel argmax categorical, outputs ----------------------
static __device__ __forceinline__ uint2 threefry2x32(uint2 key, uint2 ctr) {
  const unsigned rot[8] = {13u, 15u, 26u, 6u, 17u, 29u, 16u, 24u};
  unsigned ks[3] = {key.x, key.y, 0x1BD11BDAu ^ key.x ^ key.y};
  unsigned x0 = ctr.x + ks[0], x1 = ctr.y + ks[1];
#pragma unroll
  for (int blk = 0; blk < 5; ++blk) {
#pragma unroll
    for (int r = 0; r < 4; ++r) {
      unsigned rc = rot[(blk & 1) * 4 + r];
      x0 += x1;
      x1 = (x1 << rc) | (x1 >> (32u - rc));
      x1 ^= x0;
    }
    x0 += ks[(blk + 1) % 3];
    x1 += ks[(blk + 2) % 3] + (unsigned)(blk + 1);
  }
  return make_uint2(x0, x1);
}

__global__ __launch_bounds__(256)
void sample_kernel(const float* __restrict__ att_g, const float* __restrict__ denom,
                   const int* __restrict__ mask, float* __restrict__ out) {
  int b = blockIdx.x * blockDim.x + threadIdx.x;
  if (b >= B_N) return;
  float attv[7];
  int best = 0;
  float bestv = -1e30f;
#pragma unroll
  for (int l = 0; l < 7; ++l) {
    attv[l] = att_g[(size_t)b * 7 + l];
    float logit = (attv[l] - 10.f) - __logf(denom[l]);
    uint2 r = threefry2x32(make_uint2(0u, 42u), make_uint2((unsigned)b, (unsigned)l));
    float u = ((float)(r.x >> 8) + 0.5f) * (1.f / 16777216.f);
    float g = -__logf(-__logf(u));
    float v = logit + g;
    if (v > bestv) { bestv = v; best = l; }
  }
  float p = __expf(attv[best] - 10.f) / denom[best];
  out[b]       = (float)best;
  out[B_N + b] = p;
#pragma unroll
  for (int l = 0; l < 7; ++l)
    out[2 * B_N + (size_t)b * 7 + l] = (float)(mask[(size_t)b * 7 + l] - (l == best ? 1 : 0));
}

// ---- host side --------------------------------------------------------------
extern "C" void kernel_launch(void* const* d_in, const int* in_sizes, int n_in,
                              void* d_out, int out_size, void* d_ws, size_t ws_size,
                              hipStream_t stream) {
  (void)in_sizes; (void)n_in; (void)out_size; (void)ws_size;
  const float*     enc    = (const float*)d_in[0];
  const long long* xes    = (const long long*)d_in[1];
  const int*       mask   = (const int*)d_in[2];
  const float*     W_h    = (const float*)d_in[3];
  const float*     W_v    = (const float*)d_in[4];
  const float*     Ws_h   = (const float*)d_in[5];
  const float*     Ws_v   = (const float*)d_in[6];
  const float*     W_edge = (const float*)d_in[7];
  const float*     W_in   = (const float*)d_in[8];
  const float*     b_in   = (const float*)d_in[9];
  const float*     W_ctx  = (const float*)d_in[10];
  const float*     b_ctx  = (const float*)d_in[11];
  const float*     V      = (const float*)d_in[12];
  float* out = (float*)d_out;

  char* ws = (char*)d_ws;
  _Float16* packSel  = (_Float16*)(ws + 0);                       // 512 KB
  _Float16* packEdge = (_Float16*)(ws + (512ull << 10));          // 128 KB
  _Float16* packIn   = (_Float16*)(ws + (640ull << 10));          // 128 KB
  _Float16* packCtx  = (_Float16*)(ws + (768ull << 10));          // 128 KB
  float*    inp_g    = (float*)(ws + (1ull << 20));               // 32 MB
  float*    att_g    = (float*)(ws + (33ull << 20));              // ~0.9 MB
  float*    denom    = (float*)(ws + (34ull << 20));              // 32 B
  _Float16* enc16    = (_Float16*)(ws + (35ull << 20));           // 112 MB

  // f32 -> f16 encoder copy (58.7M elems), then pack weights (f16 B-fragments)
  cvt_kernel<<<4096, 256, 0, stream>>>(enc, enc16, (B_N * L_N * 256) / 4);
  pack_b_kernel<<<16, 256, 0, stream>>>(packSel, W_h,  32, 0);
  pack_b_kernel<<<16, 256, 0, stream>>>(packSel, W_v,  32, 8);
  pack_b_kernel<<<16, 256, 0, stream>>>(packSel, Ws_h, 32, 16);
  pack_b_kernel<<<16, 256, 0, stream>>>(packSel, Ws_v, 32, 24);
  pack_b_kernel<<<16, 256, 0, stream>>>(packEdge, W_edge, 8, 0);
  pack_b_kernel<<<16, 256, 0, stream>>>(packIn,   W_in,   8, 0);
  pack_b_kernel<<<16, 256, 0, stream>>>(packCtx,  W_ctx,  8, 0);

  scan_kernel<<<B_N / 16, 128, 0, stream>>>(enc16, xes, packSel, packEdge, packIn, b_in, inp_g);
  attn_kernel<<<(B_N * L_N) / 16 / 4, 128, 0, stream>>>(enc16, packCtx, inp_g, b_ctx, V, mask, att_g);
  zero_denom<<<1, 32, 0, stream>>>(denom);
  denom_kernel<<<64, 256, 0, stream>>>(att_g, denom);
  sample_kernel<<<(B_N + 255) / 256, 256, 0, stream>>>(att_g, denom, mask, out);
}
